// GMMGCNLayer_45509473468642
// MI455X (gfx1250) — matrix operand order repaired and checked by reading, not compiled, not run on hardware
//
#include <hip/hip_runtime.h>
#include <stdint.h>

// Problem sizes (fixed by the reference)
#define NN 4096
#define FF 128
#define OO 128
#define KK 4

typedef __attribute__((ext_vector_type(16))) __bf16   v16bf;
typedef __attribute__((ext_vector_type(8)))  float    v8f;
typedef __attribute__((ext_vector_type(4)))  uint32_t u32x4;

union Frag {
  v16bf v;
  uint32_t u[8];
  u32x4 q[2];
};

__device__ __forceinline__ uint32_t f32_to_bf16_bits(float x) {
  union { float f; uint32_t u; } c; c.f = x;
  uint32_t r = (c.u + 0x7fffu + ((c.u >> 16) & 1u)) >> 16;  // round-to-nearest-even
  return r & 0xffffu;
}
__device__ __forceinline__ float bf16_bits_to_f32(uint32_t h) {
  union { uint32_t u; float f; } c; c.u = h << 16; return c.f;
}
__device__ __forceinline__ uint32_t pack_hi_pair(float a, float b) {
  return f32_to_bf16_bits(a) | (f32_to_bf16_bits(b) << 16);
}
__device__ __forceinline__ uint32_t pack_lo_pair(float a, float b) {
  float ra = a - bf16_bits_to_f32(f32_to_bf16_bits(a));
  float rb = b - bf16_bits_to_f32(f32_to_bf16_bits(b));
  return f32_to_bf16_bits(ra) | (f32_to_bf16_bits(rb) << 16);
}

__device__ __forceinline__ v8f wmma_bf16(v16bf a, v16bf b, v8f c) {
  // v_wmma_f32_16x16x32_bf16: (neg_a, A, neg_b, B, c_mod, C, reuse_a, reuse_b)
  return __builtin_amdgcn_wmma_f32_16x16x32_bf16(false, a, false, b, (short)0, c, false, false);
}

// E[relu(X)], X ~ N(mu_t, sig_t); exact relu fallback when variance <= 0
__device__ __forceinline__ float ex_relu_f(float mu_t, float sig_t) {
  if (!(sig_t > 0.f)) return fmaxf(mu_t, 0.f);
  float ss = sqrtf(sig_t);
  float w = mu_t / ss;
  const float inv_sqrt_2pi = 0.3989422804014327f;
  return ss * (expf(-0.5f * w * w) * inv_sqrt_2pi +
               0.5f * w * (1.f + erff(w * 0.7071067811865476f)));
}

// ---------------------------------------------------------------------------
// Kernel 1: gamma[k][n] = softmax_k(pi[k] - 0.5 * sum_f valid (x-mu)^2/var)
// (k-constant terms of log N cancel in the softmax)
// ---------------------------------------------------------------------------
__global__ __launch_bounds__(128) void gamma_kernel(
    const float* __restrict__ feats, const float* __restrict__ pi,
    const float* __restrict__ mu, const float* __restrict__ sigma,
    float* __restrict__ gamma) {
  __shared__ float red[KK][128];
  int n = blockIdx.x;
  int f = threadIdx.x;
  float x = feats[(size_t)n * FF + f];
  bool isn = (x != x);
#pragma unroll
  for (int k = 0; k < KK; ++k) {
    float d = x - mu[k * FF + f];
    red[k][f] = isn ? 0.f : d * d * expf(-sigma[k * FF + f]);
  }
  __syncthreads();
  for (int st = 64; st > 0; st >>= 1) {
    if (f < st) {
#pragma unroll
      for (int k = 0; k < KK; ++k) red[k][f] += red[k][f + st];
    }
    __syncthreads();
  }
  if (f == 0) {
    float l[KK], m = -1e30f;
#pragma unroll
    for (int k = 0; k < KK; ++k) { l[k] = pi[k] - 0.5f * red[k][0]; m = fmaxf(m, l[k]); }
    float e[KK], s = 0.f;
#pragma unroll
    for (int k = 0; k < KK; ++k) { e[k] = expf(l[k] - m); s += e[k]; }
#pragma unroll
    for (int k = 0; k < KK; ++k) gamma[(size_t)k * NN + n] = e[k] / s;
  }
}

// ---------------------------------------------------------------------------
// Kernel 2: T_x[k] = mean_mat[k] @ W, T_c[k] = var_mat[k] @ (W*W), exact f32.
// Output layout is FRAGMENT-MAJOR so kernel 3 loads each WMMA B-fragment with
// two global_load_b128:
//   idx = (((k*128 + mb)*128) + o)*16 + pp
// where mb = m-block (32 rows), pp = pair-row within block (0..15),
// dword pp packs bf16 of rows (2*pp, 2*pp+1) of the m-block.
// Consumer lane (half h) reads dwords pp = h*8 .. h*8+7 contiguously.
// Each block covers 16 rows = one h-half of one m-block -> pp = hbase + (0..7).
// Each thread owns 4 consecutive pp values -> 16-byte stores.
// ---------------------------------------------------------------------------
__global__ __launch_bounds__(256) void transform_kernel(
    const float* __restrict__ feats, const float* __restrict__ weight,
    const float* __restrict__ mu, const float* __restrict__ sigma,
    uint32_t* __restrict__ txhi, uint32_t* __restrict__ txlo,
    uint32_t* __restrict__ tchi, uint32_t* __restrict__ tclo) {
  __shared__ float flds[16 * FF];
  __shared__ float mus[KK * FF];
  __shared__ float vars[KK * FF];
  int tid = threadIdx.x;
  int n0 = blockIdx.x * 16;
  int mb = n0 / 32;
  int hbase = (n0 / 2) % 16;  // 0 or 8
  for (int i = tid; i < 16 * FF; i += 256) flds[i] = feats[(size_t)n0 * FF + i];
  for (int i = tid; i < KK * FF; i += 256) { mus[i] = mu[i]; vars[i] = expf(sigma[i]); }
  __syncthreads();

  int o = tid & 127;
  int half = tid >> 7;  // thread owns local pairs p = half*4 + i (i=0..3)

  float ax[4][KK][2], ac[4][KK][2];
#pragma unroll
  for (int i = 0; i < 4; ++i)
#pragma unroll
    for (int k = 0; k < KK; ++k) {
      ax[i][k][0] = ax[i][k][1] = 0.f;
      ac[i][k][0] = ac[i][k][1] = 0.f;
    }

  for (int f = 0; f < FF; ++f) {
    float w = weight[f * OO + o];
    float wsq = w * w;
#pragma unroll
    for (int i = 0; i < 4; ++i) {
      int rr = 2 * (half * 4 + i);
      float x0 = flds[rr * FF + f];
      float x1 = flds[(rr + 1) * FF + f];
      bool i0 = (x0 != x0), i1 = (x1 != x1);
#pragma unroll
      for (int k = 0; k < KK; ++k) {
        float m = mus[k * FF + f], v = vars[k * FF + f];
        ax[i][k][0] = fmaf(i0 ? m : x0, w, ax[i][k][0]);
        ax[i][k][1] = fmaf(i1 ? m : x1, w, ax[i][k][1]);
        ac[i][k][0] = fmaf(i0 ? v : 0.f, wsq, ac[i][k][0]);
        ac[i][k][1] = fmaf(i1 ? v : 0.f, wsq, ac[i][k][1]);
      }
    }
  }

#pragma unroll
  for (int k = 0; k < KK; ++k) {
    size_t idx = (((size_t)k * 128 + mb) * 128 + o) * 16 + hbase + half * 4;
    u32x4 xh, xl, ch, cl;
#pragma unroll
    for (int i = 0; i < 4; ++i) {
      xh[i] = pack_hi_pair(ax[i][k][0], ax[i][k][1]);
      xl[i] = pack_lo_pair(ax[i][k][0], ax[i][k][1]);
      ch[i] = pack_hi_pair(ac[i][k][0], ac[i][k][1]);
      cl[i] = pack_lo_pair(ac[i][k][0], ac[i][k][1]);
    }
    *(u32x4*)&txhi[idx] = xh;
    *(u32x4*)&txlo[idx] = xl;
    *(u32x4*)&tchi[idx] = ch;
    *(u32x4*)&tclo[idx] = cl;
  }
}

// ---------------------------------------------------------------------------
// Kernel 3: fused K-component graph propagation + ex_relu + gamma mixdown.
// Workgroup = 32 output rows x 64 cols; 8 waves = 4 col-waves x 2 row-waves.
// Paired blocks (blockIdx>>1 equal) share the same shift/A2 row panel -> the
// second reader hits L2; T-fragment L2 traffic halves vs a 16-row tile.
// A tiles (shift/A2, 32x32 f32) are cooperatively converted to packed bf16
// hi/lo ONCE per workgroup into LDS (fragment layout, double-buffered); each
// wave reads its 4 A-fragments with ds_load_b128. B fragments come straight
// from the fragment-major workspace via global_load_b128 (row-wave duplicates
// hit the WGP cache). 3-product bf16 split (hi*hi + hi*lo + lo*hi) gives
// ~fp32 accuracy; shift/A2 are streamed from HBM essentially once.
// ---------------------------------------------------------------------------
__global__ __launch_bounds__(256) void conv_kernel(
    const float* __restrict__ shift, const float* __restrict__ a2,
    const uint32_t* __restrict__ txhi, const uint32_t* __restrict__ txlo,
    const uint32_t* __restrict__ tchi, const uint32_t* __restrict__ tclo,
    const float* __restrict__ gamma, float* __restrict__ out) {
  // A-fragment staging: [buffer][rowHalf][arr][lane*8+j] dwords,
  // arr: 0=Shi 1=Slo 2=Ahi 3=Alo
  __shared__ __align__(16) uint32_t abuf[2][2][4 * 256];

  int tid = threadIdx.x;
  int wave = tid >> 5;
  int lane = tid & 31;
  int rowWave = wave >> 2;              // 0..1
  int colWave = wave & 3;               // 0..3
  int n0 = (blockIdx.x >> 1) * 32;      // 32-row panel
  int colHalf = blockIdx.x & 1;
  int c0 = colHalf * 64 + colWave * 16;
  int laneLow = lane & 15;
  bool hiHalf = (lane >= 16);

  // Producer mapping: thread t converts tile rows (t>>4) and (t>>4)+16,
  // K-pair q=(t&15) -> fragment slot (lane pl, dword pj)
  int prow = tid >> 4;                  // 0..15
  int q = tid & 15;
  int pl = prow + ((q >> 3) << 4);
  int pj = q & 7;
  int pidx = pl * 8 + pj;
  const float* srow0 = shift + (size_t)(n0 + prow) * NN + 2 * q;
  const float* srow1 = srow0 + (size_t)16 * NN;
  const float* crow0 = a2 + (size_t)(n0 + prow) * NN + 2 * q;
  const float* crow1 = crow0 + (size_t)16 * NN;

  const v8f vzero = {0.f, 0.f, 0.f, 0.f, 0.f, 0.f, 0.f, 0.f};
  v8f accx[KK], accc[KK];
#pragma unroll
  for (int k = 0; k < KK; ++k) { accx[k] = vzero; accc[k] = vzero; }

  // Prologue: stage m-block 0
  {
    float2 sa = *(const float2*)(srow0);
    float2 sb = *(const float2*)(srow1);
    float2 ca = *(const float2*)(crow0);
    float2 cb = *(const float2*)(crow1);
    abuf[0][0][0 * 256 + pidx] = pack_hi_pair(sa.x, sa.y);
    abuf[0][0][1 * 256 + pidx] = pack_lo_pair(sa.x, sa.y);
    abuf[0][0][2 * 256 + pidx] = pack_hi_pair(ca.x, ca.y);
    abuf[0][0][3 * 256 + pidx] = pack_lo_pair(ca.x, ca.y);
    abuf[0][1][0 * 256 + pidx] = pack_hi_pair(sb.x, sb.y);
    abuf[0][1][1 * 256 + pidx] = pack_lo_pair(sb.x, sb.y);
    abuf[0][1][2 * 256 + pidx] = pack_hi_pair(cb.x, cb.y);
    abuf[0][1][3 * 256 + pidx] = pack_lo_pair(cb.x, cb.y);
  }

  for (int mb = 0; mb < NN / 32; ++mb) {
    __syncthreads();
    int cur = mb & 1, nxt = cur ^ 1;
    int m0n = (mb + 1) * 32;

    // Issue next tile's global loads early (overlap with WMMAs below)
    float2 sna, snb, cna, cnb;
    bool more = (m0n < NN);
    if (more) {
      sna = *(const float2*)(srow0 + m0n);
      snb = *(const float2*)(srow1 + m0n);
      cna = *(const float2*)(crow0 + m0n);
      cnb = *(const float2*)(crow1 + m0n);
      __builtin_prefetch(srow0 + m0n + 64, 0, 1);
      __builtin_prefetch(crow0 + m0n + 64, 0, 1);
    }

    // Read this step's A fragments from LDS (2x ds_load_b128 each)
    Frag shi, slo, ahi, alo;
    const uint32_t* ab = &abuf[cur][rowWave][0];
    shi.q[0] = *(const u32x4*)&ab[0 * 256 + lane * 8];
    shi.q[1] = *(const u32x4*)&ab[0 * 256 + lane * 8 + 4];
    slo.q[0] = *(const u32x4*)&ab[1 * 256 + lane * 8];
    slo.q[1] = *(const u32x4*)&ab[1 * 256 + lane * 8 + 4];
    ahi.q[0] = *(const u32x4*)&ab[2 * 256 + lane * 8];
    ahi.q[1] = *(const u32x4*)&ab[2 * 256 + lane * 8 + 4];
    alo.q[0] = *(const u32x4*)&ab[3 * 256 + lane * 8];
    alo.q[1] = *(const u32x4*)&ab[3 * 256 + lane * 8 + 4];

    // B fragments: fragment-major layout -> 2x global_load_b128 per fragment
#pragma unroll
    for (int k = 0; k < KK; ++k) {
      size_t base = (((size_t)k * 128 + mb) * 128 + c0 + laneLow) * 16 + (hiHalf ? 8 : 0);
      Frag bh, bl;
      bh.q[0] = *(const u32x4*)&txhi[base];
      bh.q[1] = *(const u32x4*)&txhi[base + 4];
      bl.q[0] = *(const u32x4*)&txlo[base];
      bl.q[1] = *(const u32x4*)&txlo[base + 4];
      accx[k] = wmma_bf16(slo.v, bh.v, accx[k]);
      accx[k] = wmma_bf16(shi.v, bl.v, accx[k]);
      accx[k] = wmma_bf16(shi.v, bh.v, accx[k]);
      bh.q[0] = *(const u32x4*)&tchi[base];
      bh.q[1] = *(const u32x4*)&tchi[base + 4];
      bl.q[0] = *(const u32x4*)&tclo[base];
      bl.q[1] = *(const u32x4*)&tclo[base + 4];
      accc[k] = wmma_bf16(alo.v, bh.v, accc[k]);
      accc[k] = wmma_bf16(ahi.v, bl.v, accc[k]);
      accc[k] = wmma_bf16(ahi.v, bh.v, accc[k]);
    }

    // Stage next tile into the other buffer
    if (more) {
      abuf[nxt][0][0 * 256 + pidx] = pack_hi_pair(sna.x, sna.y);
      abuf[nxt][0][1 * 256 + pidx] = pack_lo_pair(sna.x, sna.y);
      abuf[nxt][0][2 * 256 + pidx] = pack_hi_pair(cna.x, cna.y);
      abuf[nxt][0][3 * 256 + pidx] = pack_lo_pair(cna.x, cna.y);
      abuf[nxt][1][0 * 256 + pidx] = pack_hi_pair(snb.x, snb.y);
      abuf[nxt][1][1 * 256 + pidx] = pack_lo_pair(snb.x, snb.y);
      abuf[nxt][1][2 * 256 + pidx] = pack_hi_pair(cnb.x, cnb.y);
      abuf[nxt][1][3 * 256 + pidx] = pack_lo_pair(cnb.x, cnb.y);
    }
  }

  // Epilogue: C/D 16x16 f32 layout -> VGPR r, lanes<16: M=r, lanes>=16: M=r+8
#pragma unroll
  for (int r = 0; r < 8; ++r) {
    int n = n0 + rowWave * 16 + r + (hiHalf ? 8 : 0);
    float acc = 0.f;
#pragma unroll
    for (int k = 0; k < KK; ++k) {
      float g = gamma[(size_t)k * NN + n];
      acc += g * ex_relu_f(accx[k][r], accc[k][r]);
    }
    out[(size_t)n * OO + c0 + laneLow] = acc;
  }
}

// ---------------------------------------------------------------------------
extern "C" void kernel_launch(void* const* d_in, const int* in_sizes, int n_in,
                              void* d_out, int out_size, void* d_ws, size_t ws_size,
                              hipStream_t stream) {
  (void)in_sizes; (void)n_in; (void)out_size; (void)ws_size;
  const float* shift  = (const float*)d_in[0];
  const float* feats  = (const float*)d_in[1];
  const float* weight = (const float*)d_in[2];
  const float* pi     = (const float*)d_in[3];
  const float* mu     = (const float*)d_in[4];
  const float* sigma  = (const float*)d_in[5];
  const float* a2     = (const float*)d_in[6];
  float* out = (float*)d_out;

  // Workspace: gamma (64KB) + 4 fragment-major bf16 T arrays (4MB each) ~= 16.1MB
  char* ws = (char*)d_ws;
  float* gamma = (float*)ws;
  size_t off = ((size_t)KK * NN * sizeof(float) + 255) & ~(size_t)255;
  const size_t tbytes = (size_t)KK * (NN / 2) * OO * sizeof(uint32_t);
  uint32_t* txhi = (uint32_t*)(ws + off); off += tbytes;
  uint32_t* txlo = (uint32_t*)(ws + off); off += tbytes;
  uint32_t* tchi = (uint32_t*)(ws + off); off += tbytes;
  uint32_t* tclo = (uint32_t*)(ws + off); off += tbytes;

  gamma_kernel<<<NN, 128, 0, stream>>>(feats, pi, mu, sigma, gamma);
  transform_kernel<<<NN / 16, 256, 0, stream>>>(feats, weight, mu, sigma,
                                                txhi, txlo, tchi, tclo);
  conv_kernel<<<(NN / 32) * 2, 256, 0, stream>>>(shift, a2, txhi, txlo, tchi, tclo,
                                                 gamma, out);
}
